// EpiSEIRCNNRNNRes_PINN_82162724372700
// MI455X (gfx1250) — matrix operand order repaired and checked by reading, not compiled, not run on hardware
//
#include <hip/hip_runtime.h>
#include <hip/hip_bf16.h>

// B=32, W=64, N=256, H=8, CH=64
#define Bx 32
#define Wt 64
#define Nn 256
#define Hh 8
#define CHn 64
#define KMLP 16448           // W*N + CH
#define NMLP 2048            // H*N

typedef _Float16 v16h __attribute__((ext_vector_type(16)));
typedef float    v8f  __attribute__((ext_vector_type(8)));

// ---------------- WMMA fragment helpers ----------------
// A-matrix 16x32 f16 layout: lanes 0-15 hold K {0..7, 16..23}; lanes 16-31 hold K {8..15, 24..31}
__device__ __forceinline__ v16h packA(const float* __restrict__ row, int kb) {
  const float4* p = (const float4*)(row + kb);
  float4 f0 = p[0], f1 = p[1], f2 = p[4], f3 = p[5];
  v16h r;
  r[0]=(_Float16)f0.x; r[1]=(_Float16)f0.y; r[2]=(_Float16)f0.z; r[3]=(_Float16)f0.w;
  r[4]=(_Float16)f1.x; r[5]=(_Float16)f1.y; r[6]=(_Float16)f1.z; r[7]=(_Float16)f1.w;
  r[8]=(_Float16)f2.x; r[9]=(_Float16)f2.y; r[10]=(_Float16)f2.z; r[11]=(_Float16)f2.w;
  r[12]=(_Float16)f3.x; r[13]=(_Float16)f3.y; r[14]=(_Float16)f3.z; r[15]=(_Float16)f3.w;
  return r;
}
// B-matrix 32x16 f16 layout: lanes 0-15 hold K {0..7, 16..23}; lanes 16-31 hold K {8..15, 24..31}
__device__ __forceinline__ v16h packB(const float* __restrict__ row, int kb) {
  const float4* p = (const float4*)(row + kb);
  float4 f0 = p[0], f1 = p[1], f2 = p[4], f3 = p[5];
  v16h r;
  r[0]=(_Float16)f0.x; r[1]=(_Float16)f0.y; r[2]=(_Float16)f0.z; r[3]=(_Float16)f0.w;
  r[4]=(_Float16)f1.x; r[5]=(_Float16)f1.y; r[6]=(_Float16)f1.z; r[7]=(_Float16)f1.w;
  r[8]=(_Float16)f2.x; r[9]=(_Float16)f2.y; r[10]=(_Float16)f2.z; r[11]=(_Float16)f2.w;
  r[12]=(_Float16)f3.x; r[13]=(_Float16)f3.y; r[14]=(_Float16)f3.z; r[15]=(_Float16)f3.w;
  return r;
}
#define WMMA_F16(a,b,c) __builtin_amdgcn_wmma_f32_16x16x32_f16(false,(a),false,(b),(short)0,(c),false,false)

// ---------------- init: zero stats & h_cnn accum, bias-init gi and I_pred_dl, E_sim[:,0]=0 ----------
__global__ void k_init(float* __restrict__ stats, float* __restrict__ mlp_in,
                       float* __restrict__ gi, const float* __restrict__ bih,
                       const float* __restrict__ mlp_b, float* __restrict__ out) {
  int idx = blockIdx.x * 256 + threadIdx.x;
  if (idx < 256) { stats[idx] = 0.f; return; }
  idx -= 256;
  if (idx < Bx * KMLP) { mlp_in[idx] = 0.f; return; }
  idx -= Bx * KMLP;
  if (idx < 2048 * 192) { gi[idx] = bih[idx % 192]; return; }
  idx -= 2048 * 192;
  if (idx < Bx * NMLP) { out[idx] = mlp_b[idx & (NMLP - 1)]; return; }
  idx -= Bx * NMLP;
  if (idx < Bx * Nn) { out[131072 + (idx >> 8) * (Wt * Nn) + (idx & 255)] = 0.f; }
}

// ---------------- repack x (B,W,N) -> xg (W*B, N) for the gi GEMM ----------------
__global__ void k_xg(const float* __restrict__ x, float* __restrict__ xg) {
  int idx = blockIdx.x * 256 + threadIdx.x;
  if (idx >= Wt * Bx * Nn) return;
  int n = idx & 255, r = idx >> 8;
  int t = r >> 5, b = r & 31;
  xg[idx] = x[b * (Wt * Nn) + t * Nn + n];
}

// ---------------- Pi = softmax(pi_logits, axis=1); also pack hi/lo f16 B-fragments ----------------
__global__ __launch_bounds__(256) void k_pi(const float* __restrict__ logits, float* __restrict__ outPi,
                                            _Float16* __restrict__ pHi, _Float16* __restrict__ pLo) {
  __shared__ float sh[256];
  int n = blockIdx.x, m = threadIdx.x;
  float v = logits[n * 256 + m];
  sh[m] = v; __syncthreads();
  for (int s = 128; s > 0; s >>= 1) { if (m < s) sh[m] = fmaxf(sh[m], sh[m + s]); __syncthreads(); }
  float mx = sh[0]; __syncthreads();
  float e = __expf(v - mx);
  sh[m] = e; __syncthreads();
  for (int s = 128; s > 0; s >>= 1) { if (m < s) sh[m] += sh[m + s]; __syncthreads(); }
  float p = e / sh[0];
  outPi[n * 256 + m] = p;
  // B-fragment position for lam = I @ Pi.T : GEMM-B row index = n, K index = m
  int nt = n >> 4, ks = m >> 5, kin = m & 31;
  int lane = (n & 15) + 16 * ((kin >> 3) & 1);
  int el = (kin & 7) + 8 * ((kin >> 4) & 1);
  int addr = ((ks * 16 + nt) * 32 + lane) * 16 + el;
  _Float16 h = (_Float16)p;
  pHi[addr] = h;
  pLo[addr] = (_Float16)(p - (float)h);
}

// ---------------- conv pass 1: per-channel sum / sumsq (bias cancels through BN) ----------------
__global__ __launch_bounds__(1024) void k_conv_stats(const float* __restrict__ x, const float* __restrict__ cw,
                                                     float* __restrict__ stats) {
  __shared__ float xs[18 * 258];
  __shared__ float red[128];
  int b = blockIdx.y, w0 = blockIdx.x * 16;
  int tid = threadIdx.x;
  for (int i = tid; i < 18 * 258; i += 1024) {
    int row = i / 258, col = i % 258;
    int gw = w0 - 1 + row, gn = col - 1;
    float v = 0.f;
    if (gw >= 0 && gw < Wt && gn >= 0 && gn < Nn) v = x[b * (Wt * Nn) + gw * Nn + gn];
    xs[i] = v;
  }
  if (tid < 128) red[tid] = 0.f;
  __syncthreads();
  int c = tid & 63, n0 = (tid >> 6) * 16;
  float k0=cw[c*9+0],k1=cw[c*9+1],k2=cw[c*9+2],k3=cw[c*9+3],k4=cw[c*9+4],
        k5=cw[c*9+5],k6=cw[c*9+6],k7=cw[c*9+7],k8=cw[c*9+8];
  float s = 0.f, sq = 0.f;
  for (int wl = 0; wl < 16; ++wl) {
    const float* r0 = &xs[wl * 258 + n0];
    const float* r1 = r0 + 258;
    const float* r2 = r1 + 258;
    float a0=r0[0],a1=r0[1],b0=r1[0],b1=r1[1],c0=r2[0],c1=r2[1];
    for (int nl = 0; nl < 16; ++nl) {
      float a2=r0[nl+2], b2=r1[nl+2], c2=r2[nl+2];
      float y = k0*a0+k1*a1+k2*a2 + k3*b0+k4*b1+k5*b2 + k6*c0+k7*c1+k8*c2;
      s += y; sq += y * y;
      a0=a1;a1=a2;b0=b1;b1=b2;c0=c1;c1=c2;
    }
  }
  atomicAdd(&red[c], s);
  atomicAdd(&red[64 + c], sq);
  __syncthreads();
  if (tid < 64) { atomicAdd(&stats[tid], red[tid]); atomicAdd(&stats[64 + tid], red[64 + tid]); }
}

__global__ void k_bn_finalize(const float* __restrict__ stats, const float* __restrict__ g,
                              const float* __restrict__ bb, float* __restrict__ scsh) {
  int c = threadIdx.x;
  if (c >= 64) return;
  const float inv = 1.0f / (float)(Bx * Wt * Nn);
  float mean = stats[c] * inv;
  float var = stats[64 + c] * inv - mean * mean;
  float sc = g[c] * rsqrtf(var + 1e-5f);
  scsh[c] = sc;
  scsh[64 + c] = bb[c] - mean * sc;
}

// ---------------- conv pass 2: normalize+relu, mean over W, accumulate h_cnn into mlp_in ----------------
__global__ __launch_bounds__(1024) void k_conv_apply(const float* __restrict__ x, const float* __restrict__ cw,
                                                     const float* __restrict__ scsh, float* __restrict__ mlp_in) {
  __shared__ float xs[18 * 258];
  int b = blockIdx.y, w0 = blockIdx.x * 16;
  int tid = threadIdx.x;
  for (int i = tid; i < 18 * 258; i += 1024) {
    int row = i / 258, col = i % 258;
    int gw = w0 - 1 + row, gn = col - 1;
    float v = 0.f;
    if (gw >= 0 && gw < Wt && gn >= 0 && gn < Nn) v = x[b * (Wt * Nn) + gw * Nn + gn];
    xs[i] = v;
  }
  __syncthreads();
  int c = tid & 63, n0 = (tid >> 6) * 16;
  float k0=cw[c*9+0],k1=cw[c*9+1],k2=cw[c*9+2],k3=cw[c*9+3],k4=cw[c*9+4],
        k5=cw[c*9+5],k6=cw[c*9+6],k7=cw[c*9+7],k8=cw[c*9+8];
  float sc = scsh[c], sh = scsh[64 + c];
  float acc[16];
  for (int i = 0; i < 16; ++i) acc[i] = 0.f;
  for (int wl = 0; wl < 16; ++wl) {
    const float* r0 = &xs[wl * 258 + n0];
    const float* r1 = r0 + 258;
    const float* r2 = r1 + 258;
    float a0=r0[0],a1=r0[1],b0=r1[0],b1=r1[1],c0=r2[0],c1=r2[1];
    for (int nl = 0; nl < 16; ++nl) {
      float a2=r0[nl+2], b2=r1[nl+2], c2=r2[nl+2];
      float y = k0*a0+k1*a1+k2*a2 + k3*b0+k4*b1+k5*b2 + k6*c0+k7*c1+k8*c2;
      acc[nl] += fmaxf(y * sc + sh, 0.f);
      a0=a1;a1=a2;b0=b1;b1=b2;c0=c1;c1=c2;
    }
  }
  for (int nl = 0; nl < 16; ++nl)
    atomicAdd(&mlp_in[b * KMLP + c * Nn + n0 + nl], acc[nl] * (1.0f / (float)Wt));
}

// ---------------- generic f16-WMMA GEMM: C(MxN) += A(f32, MxK) @ Bw(f32, NxK)^T ----------------
// one wave per (mtile, 4 ntiles); grid.y splits K; f32 atomic accumulate into C.
__global__ __launch_bounds__(128) void k_gemm(const float* __restrict__ A, int lda,
                                              const float* __restrict__ Bw, int ldb,
                                              float* __restrict__ C, int ldc,
                                              int ksteps, int kchunk, int ngroups) {
  int lane = threadIdx.x & 31;
  int wv = threadIdx.x >> 5;
  int gw = blockIdx.x * 4 + wv;
  int mt = gw / ngroups, ng = gw % ngroups;
  int mBase = mt * 16, nBase0 = ng * 64;
  int ks0 = blockIdx.y * kchunk;
  int ksEnd = min(ksteps, ks0 + kchunk);
  int lhalf = (lane >= 16) ? 1 : 0;
  const float* arow = A + (long)(mBase + (lane & 15)) * lda;
  const float* brow0 = Bw + (long)(nBase0 +  0 + (lane & 15)) * ldb;
  const float* brow1 = Bw + (long)(nBase0 + 16 + (lane & 15)) * ldb;
  const float* brow2 = Bw + (long)(nBase0 + 32 + (lane & 15)) * ldb;
  const float* brow3 = Bw + (long)(nBase0 + 48 + (lane & 15)) * ldb;
  int aoff = lhalf ? 8 : 0;
  int boff = lhalf ? 8 : 0;
  v8f z = {0.f,0.f,0.f,0.f,0.f,0.f,0.f,0.f};
  v8f acc0 = z, acc1 = z, acc2 = z, acc3 = z;
  for (int ks = ks0; ks < ksEnd; ++ks) {
    int k0 = ks * 32;
    v16h af = packA(arow, k0 + aoff);
    acc0 = WMMA_F16(af, packB(brow0, k0 + boff), acc0);
    acc1 = WMMA_F16(af, packB(brow1, k0 + boff), acc1);
    acc2 = WMMA_F16(af, packB(brow2, k0 + boff), acc2);
    acc3 = WMMA_F16(af, packB(brow3, k0 + boff), acc3);
  }
  int rowB = mBase + 8 * lhalf;
  int colL = lane & 15;
  for (int j = 0; j < 8; ++j) {
    atomicAdd(&C[(long)(rowB + j) * ldc + nBase0 +  0 + colL], acc0[j]);
    atomicAdd(&C[(long)(rowB + j) * ldc + nBase0 + 16 + colL], acc1[j]);
    atomicAdd(&C[(long)(rowB + j) * ldc + nBase0 + 32 + colL], acc2[j]);
    atomicAdd(&C[(long)(rowB + j) * ldc + nBase0 + 48 + colL], acc3[j]);
  }
}

// ---------------- GRU recurrence: single workgroup, WMMA f16 for gh = h @ Whh^T ----------------
__device__ __forceinline__ void gru_frag_write(_Float16* aF, int b, int c, float h) {
  int mt = b >> 4, ks = c >> 5, kin = c & 31;
  int hl = (kin >> 3) & 1;
  int el = (kin & 7) + 8 * ((kin >> 4) & 1);
  int ln = (b & 15) + 16 * hl;
  aF[((mt * 2 + ks) * 32 + ln) * 16 + el] = (_Float16)h;
}

__global__ __launch_bounds__(1024) void k_gru(const float* __restrict__ gi, const float* __restrict__ whh,
                                              const float* __restrict__ bhh, float* __restrict__ mlp_in) {
  __shared__ float sGh[32 * 192];
  __shared__ _Float16 aF[2 * 2 * 32 * 16];
  int tid = threadIdx.x, lane = tid & 31, wv = tid >> 5;
  int b0 = tid >> 6, b1 = 16 + (tid >> 6), c = tid & 63;
  float h0 = 0.f, h1 = 0.f;
  v16h bf0 = {}, bf1 = {};
  if (wv < 12) {
    const float* brow = whh + (long)(wv * 16 + (lane & 15)) * 64;
    int kb = (lane >= 16) ? 8 : 0;
    bf0 = packB(brow, 0 + kb);
    bf1 = packB(brow, 32 + kb);
  }
  float bhr = bhh[c], bhz = bhh[64 + c], bhn = bhh[128 + c];
  gru_frag_write(aF, b0, c, 0.f);
  gru_frag_write(aF, b1, c, 0.f);
  __syncthreads();
  for (int t = 0; t < Wt; ++t) {
    if (wv < 12) {
      v8f z = {0.f,0.f,0.f,0.f,0.f,0.f,0.f,0.f};
      for (int mt = 0; mt < 2; ++mt) {
        v8f acc = z;
        v16h a0 = *(const v16h*)&aF[((mt * 2 + 0) * 32 + lane) * 16];
        acc = WMMA_F16(a0, bf0, acc);
        v16h a1 = *(const v16h*)&aF[((mt * 2 + 1) * 32 + lane) * 16];
        acc = WMMA_F16(a1, bf1, acc);
        int row = mt * 16 + 8 * (lane >= 16 ? 1 : 0);
        int col = wv * 16 + (lane & 15);
        for (int j = 0; j < 8; ++j) sGh[(row + j) * 192 + col] = acc[j];
      }
    }
    __syncthreads();
    {
      const float* gir0 = gi + (long)(t * 32 + b0) * 192;
      const float* gir1 = gi + (long)(t * 32 + b1) * 192;
      float ir0 = gir0[c], iz0 = gir0[64 + c], in0 = gir0[128 + c];
      float ir1 = gir1[c], iz1 = gir1[64 + c], in1 = gir1[128 + c];
      float hr0 = sGh[b0 * 192 + c] + bhr, hz0 = sGh[b0 * 192 + 64 + c] + bhz, hn0 = sGh[b0 * 192 + 128 + c] + bhn;
      float hr1 = sGh[b1 * 192 + c] + bhr, hz1 = sGh[b1 * 192 + 64 + c] + bhz, hn1 = sGh[b1 * 192 + 128 + c] + bhn;
      float r0 = 1.f / (1.f + __expf(-(ir0 + hr0)));
      float z0 = 1.f / (1.f + __expf(-(iz0 + hz0)));
      float n0 = tanhf(in0 + r0 * hn0);
      h0 = (1.f - z0) * n0 + z0 * h0;
      float r1 = 1.f / (1.f + __expf(-(ir1 + hr1)));
      float z1 = 1.f / (1.f + __expf(-(iz1 + hz1)));
      float n1 = tanhf(in1 + r1 * hn1);
      h1 = (1.f - z1) * n1 + z1 * h1;
      gru_frag_write(aF, b0, c, h0);
      gru_frag_write(aF, b1, c, h1);
    }
    __syncthreads();
  }
  mlp_in[b0 * KMLP + Wt * Nn + c] = h0;
  mlp_in[b1 * KMLP + Wt * Nn + c] = h1;
}

// ---------------- SEIR scan: single workgroup; lam = I @ Pi^T via hi/lo-split f16 WMMA ----------------
__device__ __forceinline__ void seir_frag_write(_Float16* aHi, _Float16* aLo, int b, int n, float v) {
  int mt = b >> 4, ks = n >> 5, kin = n & 31;
  int hl = (kin >> 3) & 1;
  int el = (kin & 7) + 8 * ((kin >> 4) & 1);
  int ln = (b & 15) + 16 * hl;
  int off = ((mt * 8 + ks) * 32 + ln) * 16 + el;
  _Float16 h = (_Float16)v;
  aHi[off] = h;
  aLo[off] = (_Float16)(v - (float)h);
}

__global__ __launch_bounds__(1024) void k_seir(const float* __restrict__ x, const float* __restrict__ beta,
                                               const float* __restrict__ sigma, const float* __restrict__ gammap,
                                               const _Float16* __restrict__ pHi, const _Float16* __restrict__ pLo,
                                               float* __restrict__ out) {
  extern __shared__ char smem[];
  float* sLam = (float*)smem;                       // 32 KB
  _Float16* aHi = (_Float16*)(smem + 32768);        // 16 KB
  _Float16* aLo = (_Float16*)(smem + 49152);        // 16 KB
  int tid = threadIdx.x, lane = tid & 31, wv = tid >> 5;
  int mt = wv >> 4, nt = wv & 15;
  float E[8], I[8], bet[8], sig[8], gam[8];
  for (int r = 0; r < 8; ++r) {
    int idx = r * 1024 + tid;
    int b = idx >> 8, n = idx & 255;
    float i0 = x[b * (Wt * Nn) + n];     // x[:,0,:]
    I[r] = i0; E[r] = 0.f;
    bet[r] = beta[n]; sig[r] = sigma[n]; gam[r] = gammap[n];
    seir_frag_write(aHi, aLo, b, n, i0);
  }
  __syncthreads();
  for (int t = 1; t < Wt; ++t) {
    v8f acc = {0.f,0.f,0.f,0.f,0.f,0.f,0.f,0.f};
    for (int ks = 0; ks < 8; ++ks) {
      int aoff = ((mt * 8 + ks) * 32 + lane) * 16;
      int boff = ((ks * 16 + nt) * 32 + lane) * 16;
      v16h ah = *(const v16h*)&aHi[aoff];
      v16h al = *(const v16h*)&aLo[aoff];
      v16h bh = *(const v16h*)&pHi[boff];
      v16h bl = *(const v16h*)&pLo[boff];
      acc = WMMA_F16(ah, bh, acc);
      acc = WMMA_F16(al, bh, acc);
      acc = WMMA_F16(ah, bl, acc);
    }
    int row = mt * 16 + 8 * (lane >= 16 ? 1 : 0);
    int col = nt * 16 + (lane & 15);
    for (int j = 0; j < 8; ++j) sLam[(row + j) * 256 + col] = acc[j];
    __syncthreads();
    for (int r = 0; r < 8; ++r) {
      int idx = r * 1024 + tid;
      int b = idx >> 8, n = idx & 255;
      float lam = sLam[b * 256 + n];
      float En = E[r] + bet[r] * lam - sig[r] * E[r];
      float In = I[r] + sig[r] * E[r] - gam[r] * I[r];
      E[r] = En; I[r] = In;
      out[131072 + b * (Wt * Nn) + t * Nn + n] = En;           // E_sim
      if (t >= Wt - Hh) out[65536 + b * (Hh * Nn) + (t - (Wt - Hh)) * Nn + n] = In;  // I_pred_pinn
      seir_frag_write(aHi, aLo, b, n, In);
    }
    __syncthreads();
  }
}

// ---------------- launcher ----------------
extern "C" void kernel_launch(void* const* d_in, const int* in_sizes, int n_in,
                              void* d_out, int out_size, void* d_ws, size_t ws_size,
                              hipStream_t stream) {
  const float* x       = (const float*)d_in[0];
  const float* conv_w  = (const float*)d_in[1];
  // conv_b (d_in[2]) cancels exactly through BN mean subtraction
  const float* bn_g    = (const float*)d_in[3];
  const float* bn_b    = (const float*)d_in[4];
  const float* gru_wih = (const float*)d_in[5];
  const float* gru_whh = (const float*)d_in[6];
  const float* gru_bih = (const float*)d_in[7];
  const float* gru_bhh = (const float*)d_in[8];
  // res_w (d_in[9]) / res_b (d_in[10]) are dead in the reference — never read
  const float* mlp_w   = (const float*)d_in[11];
  const float* mlp_b   = (const float*)d_in[12];
  const float* beta    = (const float*)d_in[13];
  const float* sigma   = (const float*)d_in[14];
  const float* gammap  = (const float*)d_in[15];
  const float* pi_log  = (const float*)d_in[16];
  float* out = (float*)d_out;

  float* ws     = (float*)d_ws;
  float* stats  = ws;                       // 128
  float* scsh   = ws + 128;                 // 128
  float* mlp_in = ws + 256;                 // 32*16448
  float* gi     = mlp_in + Bx * KMLP;       // 2048*192
  float* xg     = gi + 2048 * 192;          // 2048*256
  _Float16* pHi = (_Float16*)(xg + 2048 * 256);  // 65536 halfs
  _Float16* pLo = pHi + 65536;                    // 65536 halfs

  // init: stats=0, mlp_in=0, gi=bih, I_pred_dl=mlp_b, E_sim[:,0]=0
  {
    int total = 256 + Bx * KMLP + 2048 * 192 + Bx * NMLP + Bx * Nn;  // 993536
    k_init<<<(total + 255) / 256, 256, 0, stream>>>(stats, mlp_in, gi, gru_bih, mlp_b, out);
  }
  k_xg<<<(Wt * Bx * Nn) / 256, 256, 0, stream>>>(x, xg);
  k_pi<<<256, 256, 0, stream>>>(pi_log, out + 655360, pHi, pLo);

  k_conv_stats<<<dim3(4, 32), 1024, 0, stream>>>(x, conv_w, stats);
  k_bn_finalize<<<1, 64, 0, stream>>>(stats, bn_g, bn_b, scsh);
  k_conv_apply<<<dim3(4, 32), 1024, 0, stream>>>(x, conv_w, scsh, mlp_in);

  // gi = xg @ gru_wih^T  (M=2048, K=256, N=192): 128 mtiles * 3 ngroups / 4 waves
  k_gemm<<<dim3(96, 1), 128, 0, stream>>>(xg, 256, gru_wih, 256, gi, 192, 8, 8, 3);

  k_gru<<<1, 1024, 0, stream>>>(gi, gru_whh, gru_bhh, mlp_in);

  // I_pred_dl = mlp_in @ mlp_w^T (M=32, K=16448, N=2048): 2 mtiles * 32 ngroups / 4 waves, Ksplit=8
  k_gemm<<<dim3(16, 8), 128, 0, stream>>>(mlp_in, KMLP, mlp_w, KMLP, out, NMLP, 514, 65, 32);

  k_seir<<<1, 1024, 65536, stream>>>(x, beta, sigma, gammap, pHi, pLo, out);
}
